// FusionGNN_66958540144771
// MI455X (gfx1250) — matrix operand order repaired and checked by reading, not compiled, not run on hardware
//
#include <hip/hip_runtime.h>
#include <hip/hip_bf16.h>

// Problem constants (from the reference)
#define N_NODES 10000
#define N_EDGES 320000
#define N_GRAPH 256
#define DIM_IN  128
#define DIM_H   256
#define DIM_FP  5000
#define DIM_OUT 138

// Padded GEMM dims
#define KP_MLP1 5280          // 5256 -> pad to multiple of 32
#define NP_OUT  192           // 138  -> pad to multiple of 64

#define PANEL_ELEMS 2048      // 64 cols x 32 k-values staged per TDM copy (4 KB)

typedef __attribute__((ext_vector_type(16))) __bf16 v16bf;
typedef __attribute__((ext_vector_type(8)))  __bf16 v8bf;
typedef __attribute__((ext_vector_type(8)))  float  v8f;
typedef __attribute__((ext_vector_type(4)))  unsigned v4u;
typedef __attribute__((ext_vector_type(8)))  int      v8i;
typedef __attribute__((ext_vector_type(4)))  int      v4i;

// round-to-nearest-even fp32 -> bf16
static __device__ __forceinline__ __bf16 tobf(float f) {
  unsigned u = __builtin_bit_cast(unsigned, f);
  unsigned r = (u + 0x7fffu + ((u >> 16) & 1u)) >> 16;
  return __builtin_bit_cast(__bf16, (unsigned short)r);
}

// ---------------------------------------------------------------------------
// TDM: 1-D async copy of nElems bf16 from global to LDS (ISA ch.8 D# layout).
// group0: count=1 | lds_addr | global_addr[56:0] | type=2
// group1: data_size=1(2B) | tensor_dim0=tile_dim0=nElems | dim0_stride=nElems
// ---------------------------------------------------------------------------
static __device__ __forceinline__ void tdm_load_panel(const __bf16* gsrc,
                                                      unsigned lds_off, int nElems) {
  unsigned long long ga = (unsigned long long)gsrc;
  v4u g0 = { 1u,                                              // count=1
             lds_off,                                         // lds_addr (bytes)
             (unsigned)(ga & 0xffffffffu),                    // global_addr lo
             (unsigned)((ga >> 32) & 0x01ffffffu) | (2u << 30) }; // addr hi | type=2
  v8i g1 = { (int)(1u << 16),                                 // data_size=1 (2 bytes)
             (int)((nElems & 0xffff) << 16),                  // tensor_dim0[15:0]
             (int)(((nElems >> 16) & 0xffff) | (1 << 16)),    // dim0 hi | tensor_dim1=1
             (int)((nElems & 0xffff) << 16),                  // tile_dim0
             0,
             nElems,                                          // tensor_dim0_stride lo
             0, 0 };
  v4i z4 = { 0, 0, 0, 0 };
#if __has_include(<hip/amd_detail/amd_gfx1250_TDM.h>)
  v8i z8 = { 0, 0, 0, 0, 0, 0, 0, 0 };
  __builtin_amdgcn_tensor_load_to_lds(g0, g1, z4, z4, z8, 0); // clang-23 / therock arity
#else
  __builtin_amdgcn_tensor_load_to_lds(g0, g1, z4, z4, 0);     // ROCm 7.2 / clang-22 arity
#endif
}

// ---------------------------------------------------------------------------
// Utility: zero a float buffer (grid-stride)
// ---------------------------------------------------------------------------
__global__ void zero_f32(float* __restrict__ p, long n) {
  long i = (long)blockIdx.x * blockDim.x + threadIdx.x;
  long stride = (long)gridDim.x * blockDim.x;
  for (; i < n; i += stride) p[i] = 0.0f;
}

// ---------------------------------------------------------------------------
// GCN symmetric normalization: deg (with self loop) -> dinv = rsqrt(max(deg,1))
// ---------------------------------------------------------------------------
__global__ void deg_count(const int* __restrict__ dst, float* __restrict__ deg, int E) {
  int e = blockIdx.x * blockDim.x + threadIdx.x;
  if (e < E) atomicAdd(&deg[dst[e]], 1.0f);
}

__global__ void deg_finalize(float* __restrict__ deg, int n) {
  int i = blockIdx.x * blockDim.x + threadIdx.x;
  if (i < n) {
    float d = deg[i] + 1.0f;               // + self loop
    deg[i] = rsqrtf(fmaxf(d, 1.0f));       // store dinv in place
  }
}

// ---------------------------------------------------------------------------
// Pack A (optionally concat of A1|A2) fp32 -> bf16 row-major, K zero-padded.
// ---------------------------------------------------------------------------
__global__ void pack_a_bf16(const float* __restrict__ A1, const float* __restrict__ A2,
                            __bf16* __restrict__ out, int M, int K1, int K2, int Kpad) {
  long total  = (long)M * Kpad;
  long stride = (long)gridDim.x * blockDim.x;
  for (long i = (long)blockIdx.x * blockDim.x + threadIdx.x; i < total; i += stride) {
    int row = (int)(i / Kpad);
    int k   = (int)(i - (long)row * Kpad);
    float v = 0.0f;
    if (k < K1)            v = A1[(long)row * K1 + k];
    else if (k - K1 < K2)  v = A2[(long)row * K2 + (k - K1)];
    out[i] = tobf(v);
  }
}

// ---------------------------------------------------------------------------
// Pack W[K,N] fp32 row-major -> bf16 fragment-native layout:
//   Wp[((k>>5)*Npad + col)*32 + (k&31)]
// so each lane's B fragment (16 bf16) is one contiguous 32-byte chunk and a
// 64-column k-block panel is one contiguous 4 KB region (TDM-friendly).
// ---------------------------------------------------------------------------
__global__ void pack_b_bf16(const float* __restrict__ W, __bf16* __restrict__ out,
                            int K, int N, int Kpad, int Npad) {
  long total  = (long)Kpad * Npad;
  long stride = (long)gridDim.x * blockDim.x;
  for (long i = (long)blockIdx.x * blockDim.x + threadIdx.x; i < total; i += stride) {
    int k   = (int)(i / Npad);
    int col = (int)(i - (long)k * Npad);
    float v = (k < K && col < N) ? W[(long)k * N + col] : 0.0f;
    out[((long)(k >> 5) * Npad + col) * 32 + (k & 31)] = tobf(v);
  }
}

// ---------------------------------------------------------------------------
// One GEMM k-step: 2 A loads (global b128), 4 B fragments from LDS, 4 WMMAs.
// ---------------------------------------------------------------------------
static __device__ __forceinline__ void gemm_step(const __bf16* __restrict__ A, long aoff,
                                                 int kk, int half, int l,
                                                 const __bf16* sm, v8f* c) {
  v16bf a = *(const v16bf*)(A + aoff + kk + half * 16);
  const __bf16* wb = sm + l * 32 + half * 16;
  v16bf b0 = *(const v16bf*)(wb);
  v16bf b1 = *(const v16bf*)(wb + 16 * 32);
  v16bf b2 = *(const v16bf*)(wb + 32 * 32);
  v16bf b3 = *(const v16bf*)(wb + 48 * 32);
  c[0] = __builtin_amdgcn_wmma_f32_16x16x32_bf16(false, a, false, b0, (short)0, c[0], false, false);
  c[1] = __builtin_amdgcn_wmma_f32_16x16x32_bf16(false, a, false, b1, (short)0, c[1], false, false);
  c[2] = __builtin_amdgcn_wmma_f32_16x16x32_bf16(false, a, false, b2, (short)0, c[2], false, false);
  c[3] = __builtin_amdgcn_wmma_f32_16x16x32_bf16(false, a, false, b3, (short)0, c[3], false, false);
}

// ---------------------------------------------------------------------------
// WMMA GEMM on packed bf16: C[M,Nout] = A[M,Kpad] @ W[Kpad,Npad] (+bias)(+relu)
// One wave32 per 16x64 output strip (4 N-tiles). B panels staged into LDS by
// the Tensor Data Mover, double-buffered on TENSORcnt; A streamed from global.
// Grid: (M/16, Npad/64), block 32. M%16==0, Kpad%32==0, Npad%64==0.
// ---------------------------------------------------------------------------
__global__ void wmma_gemm_packed(const __bf16* __restrict__ A, const __bf16* __restrict__ Wp,
                                 const float* __restrict__ bias, float* __restrict__ C,
                                 int Kpad, int Npad, int Nout, int relu) {
  __shared__ __align__(64) __bf16 smem[2][PANEL_ELEMS];
  const int lane = threadIdx.x & 31;
  const int half = lane >> 4;            // 0: lanes 0-15, 1: lanes 16-31
  const int l    = lane & 15;
  const long aoff = (long)(blockIdx.x * 16 + l) * Kpad;
  const unsigned lds0 = (unsigned)(unsigned long long)&smem[0][0];
  const unsigned lds1 = (unsigned)(unsigned long long)&smem[1][0];
  const long panelStride = (long)Npad * 32;              // elements per k-block
  const __bf16* wpBase = Wp + (long)blockIdx.y * 64 * 32; // + col0*32

  v8f c[4] = {{}, {}, {}, {}};

  // Prologue: stage k-block 0 into buffer 0
  tdm_load_panel(wpBase, lds0, PANEL_ELEMS);

  int kk = 0;
  for (; kk + 32 < Kpad; kk += 32) {
    const int kb  = kk >> 5;
    const int buf = kb & 1;
    // Stage next panel into the other buffer while computing this one
    tdm_load_panel(wpBase + (long)(kb + 1) * panelStride, buf ? lds0 : lds1, PANEL_ELEMS);
    __builtin_amdgcn_s_wait_tensorcnt((short)1);   // older TDM (this buf) done
    gemm_step(A, aoff, kk, half, l, &smem[buf][0], c);
  }
  __builtin_amdgcn_s_wait_tensorcnt((short)0);     // last panel done
  gemm_step(A, aoff, kk, half, l, &smem[(kk >> 5) & 1][0], c);

  // C/D layout: VGPR i -> tile row (i + 8*half), column = lane&15
  const int rbase = blockIdx.x * 16 + half * 8;
#pragma unroll
  for (int t = 0; t < 4; ++t) {
    int col = blockIdx.y * 64 + t * 16 + l;
    if (col < Nout) {
      float bi = bias ? bias[col] : 0.0f;
#pragma unroll
      for (int i = 0; i < 8; ++i) {
        float v = c[t][i] + bi;
        if (relu) v = fmaxf(v, 0.0f);
        C[(long)(rbase + i) * Nout + col] = v;
      }
    }
  }
}

// ---------------------------------------------------------------------------
// Edge scatter: agg[dst] += hw[src] * (dinv[src]*dinv[dst]); one wave per edge.
// L2-resident atomics (feature matrix ~10 MB << 192 MB L2).
// ---------------------------------------------------------------------------
__global__ void scatter_edges(const int* __restrict__ src, const int* __restrict__ dst,
                              const float* __restrict__ dinv, const float* __restrict__ hw,
                              float* __restrict__ agg, int E, int H) {
  int warp = blockIdx.x * (blockDim.x >> 5) + (threadIdx.x >> 5);
  int lane = threadIdx.x & 31;
  if (warp >= E) return;
  int s = src[warp], d = dst[warp];
  float w = dinv[s] * dinv[d];
  const float4* hp = (const float4*)(hw + (long)s * H);
  float* ap = agg + (long)d * H;
  __builtin_prefetch(hp + lane, 0, 1);   // global_prefetch_b8 on the gathered row
  for (int f4 = lane; f4 < (H >> 2); f4 += 32) {
    float4 v = hp[f4];
    atomicAdd(ap + 4 * f4 + 0, v.x * w);
    atomicAdd(ap + 4 * f4 + 1, v.y * w);
    atomicAdd(ap + 4 * f4 + 2, v.z * w);
    atomicAdd(ap + 4 * f4 + 3, v.w * w);
  }
}

// Self-loop term + bias + ReLU
__global__ void gcn_finalize(const float* __restrict__ hw, const float* __restrict__ dinv,
                             const float* __restrict__ bias, float* __restrict__ agg,
                             long n, int H) {
  long i = (long)blockIdx.x * blockDim.x + threadIdx.x;
  if (i >= n * H) return;
  long node = i / H;
  int  f    = (int)(i - node * H);
  float di  = dinv[node];
  float v   = agg[i] + hw[i] * di * di + bias[f];
  agg[i] = fmaxf(v, 0.0f);
}

// ---------------------------------------------------------------------------
// Global mean pool
// ---------------------------------------------------------------------------
__global__ void pool_accum(const float* __restrict__ h, const int* __restrict__ batch,
                           float* __restrict__ sums, float* __restrict__ cnts,
                           int n, int H) {
  int warp = blockIdx.x * (blockDim.x >> 5) + (threadIdx.x >> 5);
  int lane = threadIdx.x & 31;
  if (warp >= n) return;
  int g = batch[warp];
  const float4* hp = (const float4*)(h + (long)warp * H);
  float* sp = sums + (long)g * H;
  for (int f4 = lane; f4 < (H >> 2); f4 += 32) {
    float4 v = hp[f4];
    atomicAdd(sp + 4 * f4 + 0, v.x);
    atomicAdd(sp + 4 * f4 + 1, v.y);
    atomicAdd(sp + 4 * f4 + 2, v.z);
    atomicAdd(sp + 4 * f4 + 3, v.w);
  }
  if (lane == 0) atomicAdd(&cnts[g], 1.0f);
}

__global__ void pool_finalize(float* __restrict__ sums, const float* __restrict__ cnts,
                              int G, int H) {
  int i = blockIdx.x * blockDim.x + threadIdx.x;
  if (i >= G * H) return;
  sums[i] /= fmaxf(cnts[i / H], 1.0f);
}

// ---------------------------------------------------------------------------
// Host-side orchestration
// ---------------------------------------------------------------------------
extern "C" void kernel_launch(void* const* d_in, const int* in_sizes, int n_in,
                              void* d_out, int out_size, void* d_ws, size_t ws_size,
                              hipStream_t stream) {
  const float* x     = (const float*)d_in[0];
  const int*   ei    = (const int*)d_in[1];        // [2, E] row-major
  const int*   batch = (const int*)d_in[2];
  const float* gfp   = (const float*)d_in[3];
  const float* W1  = (const float*)d_in[4];  const float* b1  = (const float*)d_in[5];
  const float* W2  = (const float*)d_in[6];  const float* b2  = (const float*)d_in[7];
  const float* Wm1 = (const float*)d_in[8];  const float* bm1 = (const float*)d_in[9];
  const float* Wm2 = (const float*)d_in[10]; const float* bm2 = (const float*)d_in[11];
  float* out = (float*)d_out;

  const int* src = ei;            // edge_index[0]
  const int* dst = ei + N_EDGES;  // edge_index[1]

  // Workspace carve-up (all sections 64B-aligned by construction)
  char* base = (char*)d_ws;
  float* dinv = (float*)base;  base += sizeof(float) * N_NODES;
  float* hw   = (float*)base;  base += sizeof(float) * (long)N_NODES * DIM_H;
  float* agg1 = (float*)base;  base += sizeof(float) * (long)N_NODES * DIM_H;
  float* agg2 = (float*)base;  base += sizeof(float) * (long)N_NODES * DIM_H;
  float* pool = (float*)base;  base += sizeof(float) * (long)N_GRAPH * DIM_H;
  float* cnts = (float*)base;  base += sizeof(float) * N_GRAPH;
  float* hmid = (float*)base;  base += sizeof(float) * (long)N_GRAPH * DIM_H;
  __bf16* Abf = (__bf16*)base; base += sizeof(__bf16) * (long)N_NODES * DIM_H;   // max A pack
  __bf16* Wp  = (__bf16*)base; base += sizeof(__bf16) * (long)KP_MLP1 * DIM_H;   // max W pack

  // Zero the atomic accumulators
  zero_f32<<<64,   256, 0, stream>>>(dinv, N_NODES);
  zero_f32<<<2048, 256, 0, stream>>>(agg1, (long)N_NODES * DIM_H);
  zero_f32<<<2048, 256, 0, stream>>>(agg2, (long)N_NODES * DIM_H);
  zero_f32<<<256,  256, 0, stream>>>(pool, (long)N_GRAPH * DIM_H);
  zero_f32<<<4,    256, 0, stream>>>(cnts, N_GRAPH);

  // Degree normalization
  deg_count<<<(N_EDGES + 255) / 256, 256, 0, stream>>>(dst, dinv, N_EDGES);
  deg_finalize<<<(N_NODES + 255) / 256, 256, 0, stream>>>(dinv, N_NODES);

  const int edgeBlocks = N_EDGES / 8;          // 8 waves per 256-thread block
  const long NH = (long)N_NODES * DIM_H;
  const int PACK_G = 2048;

  // ---- GCN layer 1: hw = x @ W1 ----
  pack_a_bf16<<<PACK_G, 256, 0, stream>>>(x, nullptr, Abf, N_NODES, DIM_IN, 0, DIM_IN);
  pack_b_bf16<<<PACK_G, 256, 0, stream>>>(W1, Wp, DIM_IN, DIM_H, DIM_IN, DIM_H);
  {
    dim3 g(N_NODES / 16, DIM_H / 64);
    wmma_gemm_packed<<<g, 32, 0, stream>>>(Abf, Wp, nullptr, hw, DIM_IN, DIM_H, DIM_H, 0);
  }
  scatter_edges<<<edgeBlocks, 256, 0, stream>>>(src, dst, dinv, hw, agg1, N_EDGES, DIM_H);
  gcn_finalize<<<(int)((NH + 255) / 256), 256, 0, stream>>>(hw, dinv, b1, agg1, N_NODES, DIM_H);

  // ---- GCN layer 2: hw = agg1 @ W2 ----
  pack_a_bf16<<<PACK_G, 256, 0, stream>>>(agg1, nullptr, Abf, N_NODES, DIM_H, 0, DIM_H);
  pack_b_bf16<<<PACK_G, 256, 0, stream>>>(W2, Wp, DIM_H, DIM_H, DIM_H, DIM_H);
  {
    dim3 g(N_NODES / 16, DIM_H / 64);
    wmma_gemm_packed<<<g, 32, 0, stream>>>(Abf, Wp, nullptr, hw, DIM_H, DIM_H, DIM_H, 0);
  }
  scatter_edges<<<edgeBlocks, 256, 0, stream>>>(src, dst, dinv, hw, agg2, N_EDGES, DIM_H);
  gcn_finalize<<<(int)((NH + 255) / 256), 256, 0, stream>>>(hw, dinv, b2, agg2, N_NODES, DIM_H);

  // ---- Global mean pool ----
  pool_accum<<<N_NODES / 8 + 1, 256, 0, stream>>>(agg2, batch, pool, cnts, N_NODES, DIM_H);
  pool_finalize<<<(N_GRAPH * DIM_H + 255) / 256, 256, 0, stream>>>(pool, cnts, N_GRAPH, DIM_H);

  // ---- MLP layer 1: hmid = relu(concat(pool, gfp) @ Wm1 + bm1) ----
  pack_a_bf16<<<PACK_G, 256, 0, stream>>>(pool, gfp, Abf, N_GRAPH, DIM_H, DIM_FP, KP_MLP1);
  pack_b_bf16<<<PACK_G, 256, 0, stream>>>(Wm1, Wp, DIM_H + DIM_FP, DIM_H, KP_MLP1, DIM_H);
  {
    dim3 g(N_GRAPH / 16, DIM_H / 64);
    wmma_gemm_packed<<<g, 32, 0, stream>>>(Abf, Wp, bm1, hmid, KP_MLP1, DIM_H, DIM_H, 1);
  }

  // ---- MLP layer 2: out = hmid @ Wm2 + bm2 ----
  pack_a_bf16<<<PACK_G, 256, 0, stream>>>(hmid, nullptr, Abf, N_GRAPH, DIM_H, 0, DIM_H);
  pack_b_bf16<<<PACK_G, 256, 0, stream>>>(Wm2, Wp, DIM_H, DIM_OUT, DIM_H, NP_OUT);
  {
    dim3 g(N_GRAPH / 16, NP_OUT / 64);
    wmma_gemm_packed<<<g, 32, 0, stream>>>(Abf, Wp, bm2, out, DIM_H, NP_OUT, DIM_OUT, 0);
  }
}
